// MultiHeadAttention_7352984011292
// MI455X (gfx1250) — matrix operand order repaired and checked by reading, not compiled, not run on hardware
//
#include <hip/hip_runtime.h>

// ---------------------------------------------------------------------------
// MultiHeadAttention forward for MI455X (gfx1250, wave32, WMMA f32<-f16xf16)
//   B=2, S=2048, D=1024, H=16, hd=64
//   outputs: answer [2,2048,1024] fp32 ; attention [2,16,2048,2048] fp32
// ---------------------------------------------------------------------------

typedef __attribute__((ext_vector_type(16))) _Float16 v16h;
typedef __attribute__((ext_vector_type(8)))  _Float16 v8h;
typedef __attribute__((ext_vector_type(4)))  _Float16 v4h;
typedef __attribute__((ext_vector_type(2)))  _Float16 v2h;
typedef __attribute__((ext_vector_type(8)))  float    v8f;
typedef __attribute__((ext_vector_type(4)))  float    v4f;
typedef __attribute__((ext_vector_type(4)))  int      v4i;

#define S_LEN   2048
#define D_DIM   1024
#define H_NUM   16
#define HD      64
#define ROWS    4096          // B*S

// ---- CDNA5 async global->LDS path (ASYNCcnt), guarded so a missing builtin
// ---- degrades to a plain vector copy instead of breaking the build.
#if defined(__AMDGCN__) && __has_builtin(__builtin_amdgcn_global_load_async_to_lds_b128)
#define ASYNC_LDS 1
typedef v4i __attribute__((address_space(1)))* async_gptr;
typedef v4i __attribute__((address_space(3)))* async_lptr;
#endif

__device__ __forceinline__ void copy16_g2l(const _Float16* g, _Float16* l) {
#ifdef ASYNC_LDS
  __builtin_amdgcn_global_load_async_to_lds_b128(
      (async_gptr)g, (async_lptr)l, /*offset=*/0, /*cpol=*/0);
#else
  *(v8h*)l = *(const v8h*)g;
#endif
}

__device__ __forceinline__ void async_wait_all() {
#ifdef ASYNC_LDS
#if __has_builtin(__builtin_amdgcn_s_wait_asynccnt)
  __builtin_amdgcn_s_wait_asynccnt(0);
#else
  asm volatile("s_wait_asynccnt 0" ::: "memory");
#endif
#endif
}

__device__ __forceinline__ v8f wmma16x16x32(v16h a, v16h b, v8f c) {
  return __builtin_amdgcn_wmma_f32_16x16x32_f16(
      false, a, false, b, (short)0, c, false, false);
}

// A fragment: lane holds row (lane&15); K chunks {h*8..h*8+7} and {16+h*8..}
__device__ __forceinline__ v16h frag_a_h16(const _Float16* base, int pitch, int lane) {
  const int row = lane & 15, half = lane >> 4;
  const _Float16* p = base + row * pitch;
  v8h lo = *(const v8h*)(p + half * 8);
  v8h hi = *(const v8h*)(p + 16 + half * 8);
  v16h a;
#pragma unroll
  for (int i = 0; i < 8; ++i) { a[i] = lo[i]; a[i + 8] = hi[i]; }
  return a;
}

// A fragment from fp32 source (convert in-register)
__device__ __forceinline__ v16h frag_a_f32(const float* base, int pitch, int lane) {
  const int row = lane & 15, half = lane >> 4;
  const float* p = base + row * pitch;
  v4f x0 = *(const v4f*)(p + half * 8);
  v4f x1 = *(const v4f*)(p + half * 8 + 4);
  v4f y0 = *(const v4f*)(p + 16 + half * 8);
  v4f y1 = *(const v4f*)(p + 16 + half * 8 + 4);
  v16h a;
#pragma unroll
  for (int i = 0; i < 4; ++i) {
    a[i]      = (_Float16)x0[i];
    a[i + 4]  = (_Float16)x1[i];
    a[i + 8]  = (_Float16)y0[i];
    a[i + 12] = (_Float16)y1[i];
  }
  return a;
}

// B fragment from K-major storage Bt[n][k]: lane holds col (lane&15),
// K chunks {h*8..h*8+7} and {16+h*8..}
__device__ __forceinline__ v16h frag_b_h16(const _Float16* base, int pitch, int lane) {
  const int col = lane & 15, half = lane >> 4;
  const _Float16* p = base + col * pitch;
  v8h lo = *(const v8h*)(p + half * 8);
  v8h hi = *(const v8h*)(p + 16 + half * 8);
  v16h b;
#pragma unroll
  for (int i = 0; i < 8; ++i) { b[i] = lo[i]; b[i + 8] = hi[i]; }
  return b;
}

// ---------------------------------------------------------------------------
// 1) QKV projection GEMM:  dst = f16( X[4096,1024] @ W[1024,1024] + bias )
//    block tile 64x128, 8 waves (2x4), wave tile 32x32 -> 4 WMMA / K-step
//    mode 0: dst[b,h,s,hd] (Q,K)   mode 1: dst[b,h,hd,s] (V transposed)
// ---------------------------------------------------------------------------
__global__ void __launch_bounds__(256)
proj_kernel(const float* __restrict__ X, const float* __restrict__ W,
            const float* __restrict__ bias, _Float16* __restrict__ dst, int mode) {
  constexpr int BM = 64, BN = 128, KS = 32, PITCH = 48;   // 96B rows (16B aligned)
  __shared__ __align__(16) _Float16 Xs[BM * PITCH];
  __shared__ __align__(16) _Float16 Ws[BN * PITCH];

  const int tid = threadIdx.x, lane = tid & 31, wave = tid >> 5;
  const int wr = wave >> 2, wc = wave & 3;
  const int m0 = blockIdx.x * BM, n0 = blockIdx.y * BN;

  // staging maps (fixed-trip, no predication)
  const int xr = tid >> 3,  xc = (tid & 7) * 4;       // X: rows xr, xr+32; 4 cols
  const int wn = (tid & 31) * 4, wk = (tid >> 5) * 2; // W: 4 n, k pairs {wk,wk+1},{+16}

  v8f acc[2][2] = {};

  for (int kk = 0; kk < D_DIM; kk += KS) {
    // ---- issue all global loads first (wide b128)
    v4f x0 = *(const v4f*)&X[(size_t)(m0 + xr)      * D_DIM + kk + xc];
    v4f x1 = *(const v4f*)&X[(size_t)(m0 + xr + 32) * D_DIM + kk + xc];
    v4f w0 = *(const v4f*)&W[(size_t)(kk + wk)      * D_DIM + n0 + wn];
    v4f w1 = *(const v4f*)&W[(size_t)(kk + wk + 1)  * D_DIM + n0 + wn];
    v4f w2 = *(const v4f*)&W[(size_t)(kk + wk + 16) * D_DIM + n0 + wn];
    v4f w3 = *(const v4f*)&W[(size_t)(kk + wk + 17) * D_DIM + n0 + wn];

    // ---- convert + store to LDS
    v4h hx0, hx1;
#pragma unroll
    for (int j = 0; j < 4; ++j) { hx0[j] = (_Float16)x0[j]; hx1[j] = (_Float16)x1[j]; }
    *(v4h*)&Xs[xr * PITCH + xc]        = hx0;
    *(v4h*)&Xs[(xr + 32) * PITCH + xc] = hx1;
#pragma unroll
    for (int j = 0; j < 4; ++j) {          // transposed W: Ws[n][k], pack (k,k+1)
      v2h p0; p0[0] = (_Float16)w0[j]; p0[1] = (_Float16)w1[j];
      v2h p1; p1[0] = (_Float16)w2[j]; p1[1] = (_Float16)w3[j];
      *(v2h*)&Ws[(wn + j) * PITCH + wk]      = p0;
      *(v2h*)&Ws[(wn + j) * PITCH + wk + 16] = p1;
    }
    __syncthreads();

    v16h a0 = frag_a_h16(Xs + (wr * 32)      * PITCH, PITCH, lane);
    v16h a1 = frag_a_h16(Xs + (wr * 32 + 16) * PITCH, PITCH, lane);
    v16h b0 = frag_b_h16(Ws + (wc * 32)      * PITCH, PITCH, lane);
    v16h b1 = frag_b_h16(Ws + (wc * 32 + 16) * PITCH, PITCH, lane);
    acc[0][0] = wmma16x16x32(a0, b0, acc[0][0]);
    acc[0][1] = wmma16x16x32(a0, b1, acc[0][1]);
    acc[1][0] = wmma16x16x32(a1, b0, acc[1][0]);
    acc[1][1] = wmma16x16x32(a1, b1, acc[1][1]);
    __syncthreads();
  }

  const int half = lane >> 4, colL = lane & 15;
#pragma unroll
  for (int mi = 0; mi < 2; ++mi) {
#pragma unroll
    for (int ni = 0; ni < 2; ++ni) {
      const int n_g = n0 + wc * 32 + ni * 16 + colL;
      const float bb = bias[n_g];
      const int h = n_g >> 6, d = n_g & 63;
#pragma unroll
      for (int i = 0; i < 8; ++i) {
        const int m_g = m0 + wr * 32 + mi * 16 + half * 8 + i;
        const int bt = m_g >> 11, s = m_g & (S_LEN - 1);
        const float val = acc[mi][ni][i] + bb;
        size_t off;
        if (mode == 0)
          off = (((size_t)bt * H_NUM + h) * S_LEN + s) * HD + d;   // [b,h,s,hd]
        else
          off = (((size_t)bt * H_NUM + h) * HD + d) * S_LEN + s;   // [b,h,hd,s]
        dst[off] = (_Float16)val;
      }
    }
  }
}

// ---------------------------------------------------------------------------
// 2) scores = mask( Q @ K^T / 8 ) -> fp32 raw scores into attention buffer
//    block: 32(q) x 128(k), 8 waves (2x4); wave: 16x32 tile, K=64 -> 4 WMMA
// ---------------------------------------------------------------------------
__global__ void __launch_bounds__(256)
scores_kernel(const _Float16* __restrict__ qbuf, const _Float16* __restrict__ kbuf,
              const unsigned char* __restrict__ mask, float* __restrict__ attn) {
  const int tid = threadIdx.x, lane = tid & 31, wave = tid >> 5;
  const int wr = wave >> 2, wc = wave & 3;
  const int q0 = blockIdx.x * 32, k0 = blockIdx.y * 128;
  const int bh = blockIdx.z, bt = bh >> 4;

  const _Float16* qb = qbuf + (size_t)bh * S_LEN * HD;
  const _Float16* kb = kbuf + (size_t)bh * S_LEN * HD;

  v8f acc[2] = {};
#pragma unroll
  for (int kk = 0; kk < HD; kk += 32) {
    v16h a  = frag_a_h16(qb + (size_t)(q0 + wr * 16) * HD + kk, HD, lane);
    v16h b0 = frag_b_h16(kb + (size_t)(k0 + wc * 32)      * HD + kk, HD, lane);
    v16h b1 = frag_b_h16(kb + (size_t)(k0 + wc * 32 + 16) * HD + kk, HD, lane);
    acc[0] = wmma16x16x32(a, b0, acc[0]);
    acc[1] = wmma16x16x32(a, b1, acc[1]);
  }

  const int half = lane >> 4, colL = lane & 15;
  float* out = attn + (size_t)bh * S_LEN * S_LEN;
#pragma unroll
  for (int ni = 0; ni < 2; ++ni) {
    const int kcol = k0 + wc * 32 + ni * 16 + colL;
    const bool mok = mask[(size_t)bt * S_LEN + kcol] != 0;
#pragma unroll
    for (int i = 0; i < 8; ++i) {
      const int qrow = q0 + wr * 16 + half * 8 + i;
      float s = acc[ni][i] * 0.125f;            // 1/sqrt(64)
      if (!mok) s = -1.0e9f;
      out[(size_t)qrow * S_LEN + kcol] = s;
    }
  }
}

// ---------------------------------------------------------------------------
// 3) row softmax in-place (one block / row of 2048; 8 floats live in regs)
// ---------------------------------------------------------------------------
__global__ void __launch_bounds__(256)
softmax_kernel(float* __restrict__ attn) {
  __shared__ float red[256];
  v4f* row = (v4f*)(attn + (size_t)blockIdx.x * S_LEN);   // 512 x v4f
  const int tid = threadIdx.x;

  v4f x0 = row[tid], x1 = row[tid + 256];

  float m = -3.4e38f;
#pragma unroll
  for (int c = 0; c < 4; ++c) { m = fmaxf(m, x0[c]); m = fmaxf(m, x1[c]); }
  red[tid] = m; __syncthreads();
#pragma unroll
  for (int s = 128; s > 0; s >>= 1) {
    if (tid < s) red[tid] = fmaxf(red[tid], red[tid + s]);
    __syncthreads();
  }
  m = red[0]; __syncthreads();

  float sum = 0.f;
  v4f e0, e1;
#pragma unroll
  for (int c = 0; c < 4; ++c) {
    e0[c] = __expf(x0[c] - m); e1[c] = __expf(x1[c] - m);
    sum += e0[c] + e1[c];
  }
  red[tid] = sum; __syncthreads();
#pragma unroll
  for (int s = 128; s > 0; s >>= 1) {
    if (tid < s) red[tid] += red[tid + s];
    __syncthreads();
  }
  const float inv = 1.0f / red[0];

#pragma unroll
  for (int c = 0; c < 4; ++c) { e0[c] *= inv; e1[c] *= inv; }
  row[tid] = e0; row[tid + 256] = e1;
}

// ---------------------------------------------------------------------------
// 4) out0 = P @ V   (P fp32 -> f16 in-register; V^T staged in LDS via async)
//    block: 128(q) x 64(hd); wave: 16x64 tile -> 4 WMMA / K-step, K=2048
// ---------------------------------------------------------------------------
__global__ void __launch_bounds__(256)
pv_kernel(const float* __restrict__ attn, const _Float16* __restrict__ vT,
          _Float16* __restrict__ aout) {
  constexpr int PITCH = 48;
  __shared__ __align__(16) _Float16 Vs[HD * PITCH];       // 64 x 48 f16

  const int tid = threadIdx.x, lane = tid & 31, wave = tid >> 5;
  const int q0 = blockIdx.x * 128;
  const int bh = blockIdx.y, bt = bh >> 4, h = bh & 15;

  const float*    P = attn + (size_t)bh * S_LEN * S_LEN;
  const _Float16* V = vT   + (size_t)bh * HD * S_LEN;

  const int vd = tid >> 2, vc = (tid & 3) * 8;            // one b128 per thread

  v8f acc[4] = {};

  for (int kk = 0; kk < S_LEN; kk += 32) {
    copy16_g2l(&V[(size_t)vd * S_LEN + kk + vc], &Vs[vd * PITCH + vc]);
    async_wait_all();
    __syncthreads();

    v16h a = frag_a_f32(P + (size_t)(q0 + wave * 16) * S_LEN + kk, S_LEN, lane);
#pragma unroll
    for (int ni = 0; ni < 4; ++ni) {
      v16h b = frag_b_h16(Vs + (ni * 16) * PITCH, PITCH, lane);
      acc[ni] = wmma16x16x32(a, b, acc[ni]);
    }
    __syncthreads();
  }

  const int half = lane >> 4, colL = lane & 15;
#pragma unroll
  for (int ni = 0; ni < 4; ++ni) {
    const int c = h * HD + ni * 16 + colL;                // merged-head column
#pragma unroll
    for (int i = 0; i < 8; ++i) {
      const int q = q0 + wave * 16 + half * 8 + i;
      aout[(size_t)(bt * S_LEN + q) * D_DIM + c] = (_Float16)acc[ni][i];
    }
  }
}

// ---------------------------------------------------------------------------
// 5) answer = out0[4096,1024] @ Wo + bo  (fp32 output); A tile async -> LDS
// ---------------------------------------------------------------------------
__global__ void __launch_bounds__(256)
outproj_kernel(const _Float16* __restrict__ A, const float* __restrict__ W,
               const float* __restrict__ bias, float* __restrict__ answer) {
  constexpr int BM = 64, BN = 128, KS = 32, PITCH = 48;
  __shared__ __align__(16) _Float16 Xs[BM * PITCH];
  __shared__ __align__(16) _Float16 Ws[BN * PITCH];

  const int tid = threadIdx.x, lane = tid & 31, wave = tid >> 5;
  const int wr = wave >> 2, wc = wave & 3;
  const int m0 = blockIdx.x * BM, n0 = blockIdx.y * BN;

  const int xr = tid >> 2, xc = (tid & 3) * 8;            // A: one b128 per thread
  const int wn = (tid & 31) * 4, wk = (tid >> 5) * 2;

  v8f acc[2][2] = {};

  for (int kk = 0; kk < D_DIM; kk += KS) {
    copy16_g2l(&A[(size_t)(m0 + xr) * D_DIM + kk + xc], &Xs[xr * PITCH + xc]);

    v4f w0 = *(const v4f*)&W[(size_t)(kk + wk)      * D_DIM + n0 + wn];
    v4f w1 = *(const v4f*)&W[(size_t)(kk + wk + 1)  * D_DIM + n0 + wn];
    v4f w2 = *(const v4f*)&W[(size_t)(kk + wk + 16) * D_DIM + n0 + wn];
    v4f w3 = *(const v4f*)&W[(size_t)(kk + wk + 17) * D_DIM + n0 + wn];
#pragma unroll
    for (int j = 0; j < 4; ++j) {
      v2h p0; p0[0] = (_Float16)w0[j]; p0[1] = (_Float16)w1[j];
      v2h p1; p1[0] = (_Float16)w2[j]; p1[1] = (_Float16)w3[j];
      *(v2h*)&Ws[(wn + j) * PITCH + wk]      = p0;
      *(v2h*)&Ws[(wn + j) * PITCH + wk + 16] = p1;
    }
    async_wait_all();
    __syncthreads();

    v16h a0 = frag_a_h16(Xs + (wr * 32)      * PITCH, PITCH, lane);
    v16h a1 = frag_a_h16(Xs + (wr * 32 + 16) * PITCH, PITCH, lane);
    v16h b0 = frag_b_h16(Ws + (wc * 32)      * PITCH, PITCH, lane);
    v16h b1 = frag_b_h16(Ws + (wc * 32 + 16) * PITCH, PITCH, lane);
    acc[0][0] = wmma16x16x32(a0, b0, acc[0][0]);
    acc[0][1] = wmma16x16x32(a0, b1, acc[0][1]);
    acc[1][0] = wmma16x16x32(a1, b0, acc[1][0]);
    acc[1][1] = wmma16x16x32(a1, b1, acc[1][1]);
    __syncthreads();
  }

  const int half = lane >> 4, colL = lane & 15;
#pragma unroll
  for (int mi = 0; mi < 2; ++mi) {
#pragma unroll
    for (int ni = 0; ni < 2; ++ni) {
      const int n_g = n0 + wc * 32 + ni * 16 + colL;
      const float bb = bias[n_g];
#pragma unroll
      for (int i = 0; i < 8; ++i) {
        const int m_g = m0 + wr * 32 + mi * 16 + half * 8 + i;
        answer[(size_t)m_g * D_DIM + n_g] = acc[mi][ni][i] + bb;
      }
    }
  }
}

// ---------------------------------------------------------------------------
extern "C" void kernel_launch(void* const* d_in, const int* in_sizes, int n_in,
                              void* d_out, int out_size, void* d_ws, size_t ws_size,
                              hipStream_t stream) {
  (void)in_sizes; (void)n_in; (void)out_size; (void)ws_size;

  const float* query = (const float*)d_in[0];
  const float* key_  = (const float*)d_in[1];
  const float* value = (const float*)d_in[2];
  const unsigned char* mask = (const unsigned char*)d_in[3];
  const float* Wq = (const float*)d_in[4];
  const float* bq = (const float*)d_in[5];
  const float* Wk = (const float*)d_in[6];
  const float* bk = (const float*)d_in[7];
  const float* Wv = (const float*)d_in[8];
  const float* bv = (const float*)d_in[9];
  const float* Wo = (const float*)d_in[10];
  const float* bo = (const float*)d_in[11];

  float* answer = (float*)d_out;                               // 4,194,304 f32
  float* attn   = (float*)d_out + (size_t)ROWS * D_DIM;        // 134,217,728 f32

  _Float16* qbuf = (_Float16*)d_ws;                 // [b,h,s,64]  f16
  _Float16* kbuf = qbuf + (size_t)ROWS * D_DIM;     // [b,h,s,64]  f16
  _Float16* vT   = kbuf + (size_t)ROWS * D_DIM;     // [b,h,64,s]  f16
  _Float16* aout = vT   + (size_t)ROWS * D_DIM;     // [4096,1024] f16  (32 MB total)

  const dim3 blk(256);

  proj_kernel<<<dim3(ROWS / 64, D_DIM / 128), blk, 0, stream>>>(query, Wq, bq, qbuf, 0);
  proj_kernel<<<dim3(ROWS / 64, D_DIM / 128), blk, 0, stream>>>(key_,  Wk, bk, kbuf, 0);
  proj_kernel<<<dim3(ROWS / 64, D_DIM / 128), blk, 0, stream>>>(value, Wv, bv, vT,   1);

  scores_kernel<<<dim3(S_LEN / 32, S_LEN / 128, 2 * H_NUM), blk, 0, stream>>>(qbuf, kbuf, mask, attn);

  softmax_kernel<<<dim3(2 * H_NUM * S_LEN), blk, 0, stream>>>(attn);

  pv_kernel<<<dim3(S_LEN / 128, 2 * H_NUM), blk, 0, stream>>>(attn, vT, aout);

  outproj_kernel<<<dim3(ROWS / 64, D_DIM / 128), blk, 0, stream>>>(aout, Wo, bo, answer);
}
